// CRF_12979391169127
// MI455X (gfx1250) — matrix operand order repaired and checked by reading, not compiled, not run on hardware
//
#include <hip/hip_runtime.h>
#include <hip/hip_bf16.h>

typedef __attribute__((ext_vector_type(16))) _Float16 v16h;
typedef __attribute__((ext_vector_type(8)))  _Float16 v8h;
typedef __attribute__((ext_vector_type(8)))  float    v8f;

#define BB 256
#define SS 512
#define TT 32
#define START_TAG 30
#define STOP_TAG  31
#define NC 32          // chunks
#define CL 16          // steps per chunk (NC*CL == SS)

#define LOG2E 1.4426950408889634f
#define LN2   0.6931471805599453f

// Pure-VALU cross-half swap (lane l <-> l^16) via v_permlanex16_b32.
__device__ __forceinline__ float swap16(float v) {
    unsigned u = __builtin_bit_cast(unsigned, v);
    unsigned r = __builtin_amdgcn_permlanex16(u, u, 0x76543210u, 0xFEDCBA98u,
                                              false, false);
    return __builtin_bit_cast(float, r);
}

__device__ __forceinline__ float wave_max(float v) {
#pragma unroll
    for (int m = 16; m; m >>= 1) v = fmaxf(v, __shfl_xor(v, m, 32));
    return v;
}
__device__ __forceinline__ float wave_sum(float v) {
#pragma unroll
    for (int m = 16; m; m >>= 1) v += __shfl_xor(v, m, 32);
    return v;
}

// Build one WMMA B-tile (K = 32 "j" rows, N = 16 columns) from per-lane state
// hh[16] holding values for j in Jset(g) = {8g..8g+7} U {16+8g..16+8g+7}.
// B takes each lane's own hh in this k order; the swapped recv[] is unused.
__device__ __forceinline__ v16h build_b(const float* hh, int g) {
    float recv[8];
#pragma unroll
    for (int t = 0; t < 8; ++t) recv[t] = swap16(g ? hh[t] : hh[t + 8]);
    v16h Bv;
#pragma unroll
    for (int t = 0; t < 8; ++t) Bv[t]     = (_Float16)hh[t];
#pragma unroll
    for (int t = 0; t < 8; ++t) Bv[t + 8] = (_Float16)hh[t + 8];
    return Bv;
}

// ===========================================================================
// PHASE 1: per-(batch, chunk) operator product. 8192 independent waves.
// Operator kept transposed + scaled-exp: state h[n][k] = H[i][j],
//   i = 16n + (lane&15),  j = (k<8 ? 8g+k : 16+8g+(k-8)),  H in [0,1],
//   true log2-operator G[i][j] = r_i + log2 H[i][j].
// Step (unmasked): H' = normalize_rows( (H x E) * diag(exp feat_s) ),
// computed transposed as E^T x H^T via 4 WMMAs.
// ===========================================================================
__global__ __launch_bounds__(32)
void crf_chunk_ops_kernel(const float* __restrict__ feats,
                          const float* __restrict__ trans,
                          const int*   __restrict__ mask,
                          _Float16*    __restrict__ Hbuf,
                          float*       __restrict__ Rbuf)
{
    const int lane = threadIdx.x;
    const int g    = lane >> 4;
    const int bl   = lane & 15;
    const int c    = blockIdx.x;   // chunk
    const int b    = blockIdx.y;   // batch

    // constant A = E^T tiles (E = exp(trans)); A-layout: M = bl (both halves)
    v16h Alo, Ahi;
#pragma unroll
    for (int j = 0; j < 16; ++j) {
        int K = (j < 8) ? (8 * g + j) : (16 + 8 * g + (j - 8));
        Alo[j] = (_Float16)__expf(trans[K * TT + bl]);
        Ahi[j] = (_Float16)__expf(trans[K * TT + 16 + bl]);
    }

    // identity operator
    float h[2][16];
    float r0 = 0.0f, r1 = 0.0f;
#pragma unroll
    for (int n = 0; n < 2; ++n) {
        int i = 16 * n + bl;
#pragma unroll
        for (int k = 0; k < 16; ++k) {
            int j = (k < 8) ? (8 * g + k) : (16 + 8 * g + (k - 8));
            h[n][k] = (j == i) ? 1.0f : 0.0f;
        }
    }

    const v8f zero = {};
    int s_begin = c * CL; if (s_begin == 0) s_begin = 1;
    const int s_end = c * CL + CL;

    for (int s = s_begin; s < s_end; ++s) {
        int mk = __builtin_amdgcn_readfirstlane(mask[b * SS + s]);
        if (!mk) continue;                       // masked step == identity

        const float* fbase = feats + ((size_t)b * SS + s) * TT;
        const float4* f1 = (const float4*)(fbase + 8 * g);
        const float4* f2 = (const float4*)(fbase + 16 + 8 * g);
        float4 fa = f1[0], fb = f1[1], fc = f2[0], fd = f2[1];
        float fv[16] = {fa.x, fa.y, fa.z, fa.w, fb.x, fb.y, fb.z, fb.w,
                        fc.x, fc.y, fc.z, fc.w, fd.x, fd.y, fd.z, fd.w};

        v16h B0 = build_b(h[0], g);
        v16h B1 = build_b(h[1], g);

        v8f c00 = __builtin_amdgcn_wmma_f32_16x16x32_f16(
                      false, Alo, false, B0, (short)0, zero, false, false);
        v8f c10 = __builtin_amdgcn_wmma_f32_16x16x32_f16(
                      false, Ahi, false, B0, (short)0, zero, false, false);
        v8f c01 = __builtin_amdgcn_wmma_f32_16x16x32_f16(
                      false, Alo, false, B1, (short)0, zero, false, false);
        v8f c11 = __builtin_amdgcn_wmma_f32_16x16x32_f16(
                      false, Ahi, false, B1, (short)0, zero, false, false);

        float ef[16];
#pragma unroll
        for (int k = 0; k < 16; ++k)
            ef[k] = __builtin_amdgcn_exp2f(fv[k] * LOG2E);   // exp(feat_j)

        float w0[16], w1[16];
#pragma unroll
        for (int r = 0; r < 8; ++r) {
            w0[r]     = c00[r] * ef[r];
            w0[r + 8] = c10[r] * ef[r + 8];
            w1[r]     = c01[r] * ef[r];
            w1[r + 8] = c11[r] * ef[r + 8];
        }

        float m0 = w0[0], m1 = w1[0];
#pragma unroll
        for (int k = 1; k < 16; ++k) { m0 = fmaxf(m0, w0[k]); m1 = fmaxf(m1, w1[k]); }
        m0 = fmaxf(m0, swap16(m0));    // full row max (rows never all-zero)
        m1 = fmaxf(m1, swap16(m1));

        float i0 = __builtin_amdgcn_rcpf(m0);
        float i1 = __builtin_amdgcn_rcpf(m1);
        r0 += __builtin_amdgcn_logf(m0);
        r1 += __builtin_amdgcn_logf(m1);
#pragma unroll
        for (int k = 0; k < 16; ++k) { h[0][k] = w0[k] * i0; h[1][k] = w1[k] * i1; }
    }

    // store operator: Hbuf row-major H[i][j] (two contiguous 8-half runs/lane)
    const size_t base = ((size_t)b * NC + c) * (TT * TT);
#pragma unroll
    for (int n = 0; n < 2; ++n) {
        int i = 16 * n + bl;
        v8h p1, p2;
#pragma unroll
        for (int k = 0; k < 8; ++k) { p1[k] = (_Float16)h[n][k];
                                      p2[k] = (_Float16)h[n][k + 8]; }
        *(v8h*)(Hbuf + base + (size_t)i * TT + 8 * g)      = p1;
        *(v8h*)(Hbuf + base + (size_t)i * TT + 16 + 8 * g) = p2;
    }
    if (g == 0) {
        Rbuf[((size_t)b * NC + c) * TT + bl]      = r0;
        Rbuf[((size_t)b * NC + c) * TT + 16 + bl] = r1;
    }
}

// ===========================================================================
// PHASE 2: per-batch sequential combine over 32 chunk operators (lane = tag).
// q'_j = m + log2( sum_i 2^(q_i + r_i - m) * H[i][j] )
// ===========================================================================
__global__ __launch_bounds__(32)
void crf_combine_kernel(const float* __restrict__ feats,
                        const float* __restrict__ trans,
                        const _Float16* __restrict__ Hbuf,
                        const float* __restrict__ Rbuf,
                        float* __restrict__ fwd_out)
{
    __shared__ float xs[TT];
    const int lane = threadIdx.x;
    const int b    = blockIdx.x;

    float q = (feats[(size_t)b * SS * TT + lane] +
               trans[START_TAG * TT + lane]) * LOG2E;

    for (int c = 0; c < NC; ++c) {
        float r = Rbuf[((size_t)b * NC + c) * TT + lane];
        float t = q + r;
        float m = wave_max(t);
        xs[lane] = __builtin_amdgcn_exp2f(t - m);
        __syncthreads();

        const _Float16* Hp = Hbuf + ((size_t)b * NC + c) * (TT * TT) + lane;
        float s = 0.0f;
#pragma unroll
        for (int i = 0; i < TT; ++i)
            s = __builtin_fmaf(xs[i], (float)Hp[(size_t)i * TT], s);

        q = m + __builtin_amdgcn_logf(s);   // s==0 -> -inf (START column) ok
        __syncthreads();
    }

    float t = q + trans[lane * TT + STOP_TAG] * LOG2E;
    float m = wave_max(t);
    float st = wave_sum(__builtin_amdgcn_exp2f(t - m));
    if (lane == 0)
        fwd_out[b] = (m + __builtin_amdgcn_logf(st)) * LN2;
}

// ===========================================================================
// Fallback: monolithic scan (round-2 kernel), used only if ws too small.
// ===========================================================================
__global__ __launch_bounds__(32)
void crf_forward_kernel(const float* __restrict__ feats,
                        const float* __restrict__ trans,
                        const int*   __restrict__ mask,
                        float*       __restrict__ fwd_out)
{
    const int lane = threadIdx.x;
    const int g    = lane >> 4;
    const int bl   = lane & 15;
    const int b    = blockIdx.x * 16 + bl;

    v16h Alo, Ahi;
#pragma unroll
    for (int j = 0; j < 16; ++j) {
        int K = (j < 8) ? (8 * g + j) : (16 + 8 * g + (j - 8));
        Alo[j] = (_Float16)__expf(trans[K * TT + bl]);
        Ahi[j] = (_Float16)__expf(trans[K * TT + 16 + bl]);
    }

    float qj[16];
    {
        const float* f0 = feats + (size_t)b * SS * TT;
#pragma unroll
        for (int k = 0; k < 16; ++k) {
            int t = (k < 8) ? (8 * g + k) : (16 + 8 * g + (k - 8));
            qj[k] = (f0[t] + trans[START_TAG * TT + t]) * LOG2E;
        }
    }

    const v8f zero = {};
    for (int s = 1; s < SS; ++s) {
        const float* fbase = feats + ((size_t)b * SS + s) * TT;
        const float4* f1 = (const float4*)(fbase + 8 * g);
        const float4* f2 = (const float4*)(fbase + 16 + 8 * g);
        float4 fa = f1[0], fb = f1[1], fc = f2[0], fd = f2[1];
        int mk = mask[b * SS + s];

        float mloc = qj[0];
#pragma unroll
        for (int k = 1; k < 16; ++k) mloc = fmaxf(mloc, qj[k]);
        float m2 = fmaxf(mloc, swap16(mloc));

        float e[16];
#pragma unroll
        for (int k = 0; k < 16; ++k) e[k] = __builtin_amdgcn_exp2f(qj[k] - m2);

        v16h Bm = build_b(e, g);
        v8f c0 = __builtin_amdgcn_wmma_f32_16x16x32_f16(
                     false, Alo, false, Bm, (short)0, zero, false, false);
        v8f c1 = __builtin_amdgcn_wmma_f32_16x16x32_f16(
                     false, Ahi, false, Bm, (short)0, zero, false, false);

        float fv[16] = {fa.x, fa.y, fa.z, fa.w, fb.x, fb.y, fb.z, fb.w,
                        fc.x, fc.y, fc.z, fc.w, fd.x, fd.y, fd.z, fd.w};
#pragma unroll
        for (int r = 0; r < 8; ++r) {
            float n0 = __builtin_fmaf(fv[r], LOG2E,
                                      m2 + __builtin_amdgcn_logf(c0[r]));
            float n1 = __builtin_fmaf(fv[r + 8], LOG2E,
                                      m2 + __builtin_amdgcn_logf(c1[r]));
            qj[r]     = mk ? n0 : qj[r];
            qj[r + 8] = mk ? n1 : qj[r + 8];
        }
    }

    float mloc = qj[0];
#pragma unroll
    for (int k = 1; k < 16; ++k) mloc = fmaxf(mloc, qj[k]);
    float m2 = fmaxf(mloc, swap16(mloc));
    float part = 0.0f;
#pragma unroll
    for (int k = 0; k < 16; ++k) {
        int t = (k < 8) ? (8 * g + k) : (16 + 8 * g + (k - 8));
        part += __builtin_amdgcn_exp2f(qj[k] - m2) * __expf(trans[t * TT + STOP_TAG]);
    }
    float tot = part + swap16(part);
    if (lane < 16)
        fwd_out[b] = (m2 + __builtin_amdgcn_logf(tot)) * LN2;
}

// ---------------------------------------------------------------------------
__global__ __launch_bounds__(256)
void crf_gold_kernel(const float* __restrict__ feats,
                     const float* __restrict__ trans,
                     const int*   __restrict__ mask,
                     const int*   __restrict__ tags,
                     float*       __restrict__ gold_out)
{
    int b = blockIdx.x * blockDim.x + threadIdx.x;
    if (b >= BB) return;
    float acc = 0.0f;
    int prev = START_TAG, endtag = START_TAG;
    for (int s = 0; s < SS; ++s) {
        int tg = tags[b * SS + s];
        int mk = mask[b * SS + s];
        if (mk) {
            acc += feats[((size_t)b * SS + s) * TT + tg] + trans[prev * TT + tg];
            endtag = tg;
        }
        prev = tg;
    }
    gold_out[b] = acc + trans[endtag * TT + STOP_TAG];
}

__global__ __launch_bounds__(256)
void crf_reduce_kernel(const float* __restrict__ fwd,
                       const float* __restrict__ gold,
                       float*       __restrict__ out)
{
    __shared__ float sm[256];
    int t = threadIdx.x;
    sm[t] = fwd[t] - gold[t];
    __syncthreads();
#pragma unroll
    for (int off = 128; off > 0; off >>= 1) {
        if (t < off) sm[t] += sm[t + off];
        __syncthreads();
    }
    if (t == 0) out[0] = sm[0];
}

extern "C" void kernel_launch(void* const* d_in, const int* in_sizes, int n_in,
                              void* d_out, int out_size, void* d_ws, size_t ws_size,
                              hipStream_t stream)
{
    (void)in_sizes; (void)n_in; (void)out_size;

    const float* feats = (const float*)d_in[0];
    const float* trans = (const float*)d_in[1];
    const int*   mask  = (const int*)d_in[2];
    const int*   tags  = (const int*)d_in[3];

    const size_t nH = (size_t)BB * NC * TT * TT;           // f16 elements
    const size_t nR = (size_t)BB * NC * TT;                // f32 elements
    const size_t need = nH * sizeof(_Float16) + nR * sizeof(float)
                      + 2 * BB * sizeof(float);

    if (ws_size >= need) {
        _Float16* Hbuf = (_Float16*)d_ws;
        float*    Rbuf = (float*)(Hbuf + nH);
        float*    fwd  = Rbuf + nR;
        float*    gold = fwd + BB;

        crf_gold_kernel<<<1, 256, 0, stream>>>(feats, trans, mask, tags, gold);
        crf_chunk_ops_kernel<<<dim3(NC, BB), 32, 0, stream>>>(
            feats, trans, mask, Hbuf, Rbuf);
        crf_combine_kernel<<<BB, 32, 0, stream>>>(feats, trans, Hbuf, Rbuf, fwd);
        crf_reduce_kernel<<<1, 256, 0, stream>>>(fwd, gold, (float*)d_out);
    } else {
        float* fwd  = (float*)d_ws;
        float* gold = fwd + BB;
        crf_gold_kernel<<<1, 256, 0, stream>>>(feats, trans, mask, tags, gold);
        crf_forward_kernel<<<16, 32, 0, stream>>>(feats, trans, mask, fwd);
        crf_reduce_kernel<<<1, 256, 0, stream>>>(fwd, gold, (float*)d_out);
    }
}
